// LSTM_pred_38173669327431
// MI455X (gfx1250) — matrix operand order repaired and checked by reading, not compiled, not run on hardware
//
#include <hip/hip_runtime.h>

// ---------------------------------------------------------------------------
// Fused 4-layer zero-state LSTM + ELU for MI455X (gfx1250, wave32, WMMA).
//   tokens = B*S = 131072, D_IN=512, H=256, gates used: i,g,o (f is dead).
// v2: 128 tokens/block, 4-row-tile register blocking per wave so each weight
//     B-fragment feeds 4 WMMAs (4x less L2 weight traffic than v1).
// ---------------------------------------------------------------------------

typedef __bf16 bf16_t;
typedef bf16_t v16bf __attribute__((ext_vector_type(16)));
typedef float  v8f   __attribute__((ext_vector_type(8)));

#define TOK_PER_BLK 128
#define HID   256
#define DIN   512
#define NT    48          // packed gate column tiles (768 / 16)
#define KS0   16          // DIN / 32
#define KSR   8           // HID / 32
#define FRAG  512         // bf16 elements per 16x16x32 B-fragment
#define S0    (DIN + 8)   // LDS row stride (halves), padded vs bank conflicts
#define SR    (HID + 8)

__device__ __forceinline__ unsigned short f2bf(float f) {
  union { float f; unsigned u; } v; v.f = f;
  unsigned r = v.u + 0x7FFFu + ((v.u >> 16) & 1u);   // round-to-nearest-even
  return (unsigned short)(r >> 16);
}

__device__ __forceinline__ v16bf mk16(uint4 a, uint4 b) {
  union { unsigned u[8]; v16bf v; } t;
  t.u[0] = a.x; t.u[1] = a.y; t.u[2] = a.z; t.u[3] = a.w;
  t.u[4] = b.x; t.u[5] = b.y; t.u[6] = b.z; t.u[7] = b.w;
  return t.v;
}

__device__ __forceinline__ v16bf load_wfrag(const unsigned short* __restrict__ W,
                                            int frag, int lane) {
  const unsigned short* p = W + (size_t)frag * FRAG + lane * 16;  // 32B/lane, coalesced
  const uint4* q = (const uint4*)p;
  return mk16(q[0], q[1]);
}

__device__ __forceinline__ float sigm(float x)  { return 1.f / (1.f + __expf(-x)); }
__device__ __forceinline__ float tanhq(float x) { float e = __expf(2.f * x); return 1.f - 2.f / (e + 1.f); }

// ---------------------------------------------------------------------------
// Weight packers.  Fragment layout (B matrix, 32x16 K x N, bf16):
//   lane 0-15  : column n = lane,     halves h=0..15 -> K = s*32 + h
//   lane 16-31 : column n = lane-16,  halves h=0..15 -> K = s*32 + 16 + h
// Packed gate column c (0..767) -> weight row: c<256 ? c (i) : c+256 (g,o).
// ---------------------------------------------------------------------------
__global__ void pack_l0(const float* __restrict__ W0, unsigned short* __restrict__ out) {
  int idx = blockIdx.x * 256 + threadIdx.x;      // 48*16*512 exact
  int f = idx >> 9, e = idx & 511;
  int t = f >> 4, s = f & 15;
  int ln = e >> 4, h = e & 15;
  int col = t * 16 + (ln & 15);
  int row = col < 256 ? col : col + 256;
  int k   = s * 32 + ((ln >> 4) << 4) + h;
  out[idx] = f2bf(W0[row * DIN + k]);
}

__global__ void pack_rest(const float* __restrict__ Wr, unsigned short* __restrict__ out) {
  int idx = blockIdx.x * 256 + threadIdx.x;      // 3*48*8*512 exact
  int l  = idx / (NT * KSR * FRAG);
  int r0 = idx % (NT * KSR * FRAG);
  int f = r0 >> 9, e = r0 & 511;
  int t = f >> 3, s = f & 7;
  int ln = e >> 4, h = e & 15;
  int col = t * 16 + (ln & 15);
  int row = col < 256 ? col : col + 256;
  int k   = s * 32 + ((ln >> 4) << 4) + h;
  out[idx] = f2bf(Wr[(size_t)l * 1024 * HID + row * HID + k]);
}

__global__ void pack_bias(const float* __restrict__ bih, const float* __restrict__ bhh,
                          float* __restrict__ out) {
  int idx = blockIdx.x * 256 + threadIdx.x;      // 4*768 = 3072 exact
  int l = idx / 768, c = idx % 768;
  int row = c < 256 ? c : c + 256;
  out[idx] = bih[l * 1024 + row] + bhh[l * 1024 + row];
}

// ---------------------------------------------------------------------------
// One fused layer with 4-row-tile M-blocking.
// A fragment layout (16x32 M x K bf16):
//   lane 0-15  : row M=lane, halves 0..15 -> K=s*32+0..15
//   lane 16-31 : row M=lane-16, halves 0..15 -> K=s*32+16..31
// C/D layout: lane%16 = N, vgpr r -> M = r + 8*(lane>=16).
// ---------------------------------------------------------------------------
template <int KIN, int SIN, bool FINAL>
__device__ __forceinline__ void layer_pass(const unsigned short* __restrict__ inBuf,
                                           const unsigned short* __restrict__ W,
                                           const float* __restrict__ bias,
                                           unsigned short* outBuf,
                                           float* __restrict__ gOut, int tokenBase,
                                           int quad, int colQ, int lane) {
  constexpr int KS = KIN / 32;
  const int nlo = lane & 15;
  const int hi  = lane >> 4;
  const int kph = hi * 16;

  for (int c4 = 0; c4 < 4; ++c4) {
    const int ct = colQ * 4 + c4;                 // h column tile 0..15
    if (c4 < 3)
      __builtin_prefetch(W + (size_t)(ct + 1) * KS * FRAG + lane * 16, 0, 0);

    const float bi = bias[ct * 16 + nlo];
    const float bg = bias[(ct + 16) * 16 + nlo];
    const float bo = bias[(ct + 32) * 16 + nlo];
    v8f ai[4], ag[4], ao[4];
#pragma unroll
    for (int j = 0; j < 4; ++j) {
      ai[j] = (v8f){bi, bi, bi, bi, bi, bi, bi, bi};
      ag[j] = (v8f){bg, bg, bg, bg, bg, bg, bg, bg};
      ao[j] = (v8f){bo, bo, bo, bo, bo, bo, bo, bo};
    }

#pragma unroll
    for (int s = 0; s < KS; ++s) {
      const v16bf wi = load_wfrag(W, ct * KS + s, lane);
      const v16bf wg = load_wfrag(W, (ct + 16) * KS + s, lane);
      const v16bf wo = load_wfrag(W, (ct + 32) * KS + s, lane);
#pragma unroll
      for (int j = 0; j < 4; ++j) {
        const int row0 = (quad * 4 + j) * 16 + nlo;
        const unsigned short* pa = inBuf + row0 * SIN + s * 32 + kph;
        const uint4 lo = *(const uint4*)pa;
        const uint4 hh = *(const uint4*)(pa + 8);
        const v16bf a = mk16(lo, hh);
        ai[j] = __builtin_amdgcn_wmma_f32_16x16x32_bf16(false, a, false, wi,
                                                        (short)0, ai[j], false, false);
        ag[j] = __builtin_amdgcn_wmma_f32_16x16x32_bf16(false, a, false, wg,
                                                        (short)0, ag[j], false, false);
        ao[j] = __builtin_amdgcn_wmma_f32_16x16x32_bf16(false, a, false, wo,
                                                        (short)0, ao[j], false, false);
      }
    }

    const int col = ct * 16 + nlo;
#pragma unroll
    for (int j = 0; j < 4; ++j) {
      const int rbase = (quad * 4 + j) * 16 + hi * 8;
#pragma unroll
      for (int r = 0; r < 8; ++r) {
        float cc = sigm(ai[j][r]) * tanhq(ag[j][r]);
        float hv = sigm(ao[j][r]) * tanhq(cc);
        if (FINAL) {
          float e = hv > 0.f ? hv : (__expf(hv) - 1.f);   // ELU
          gOut[(size_t)(tokenBase + rbase + r) * HID + col] = e;
        } else {
          outBuf[(rbase + r) * SR + col] = f2bf(hv);
        }
      }
    }
  }
}

__global__ __launch_bounds__(256) void lstm_fused(const float* __restrict__ x,
                                                  const unsigned short* __restrict__ w0,
                                                  const unsigned short* __restrict__ wr,
                                                  const float* __restrict__ biasP,
                                                  float* __restrict__ out) {
  extern __shared__ unsigned short smem[];
  unsigned short* bufA = smem;                         // 128 x S0 halves
  unsigned short* bufB = smem + TOK_PER_BLK * S0;      // 128 x SR halves

  const int tid = threadIdx.x;
  const int lane = tid & 31;
  const int wave = tid >> 5;
  const int quad = wave & 1;      // which 4 row tiles (rows 0-63 / 64-127)
  const int colQ = wave >> 1;     // which 4 h-column tiles (cols 64*colQ..)
  const int tokenBase = blockIdx.x * TOK_PER_BLK;

  // Stage 128x512 fp32 input as bf16 into LDS (coalesced float4 reads).
  const float4* x4 = (const float4*)(x + (size_t)tokenBase * DIN);
  for (int i = tid; i < TOK_PER_BLK * DIN / 4; i += 256) {
    float4 v = x4[i];
    int r = (i * 4) / DIN, c = (i * 4) % DIN;
    unsigned short* p = bufA + r * S0 + c;
    p[0] = f2bf(v.x); p[1] = f2bf(v.y); p[2] = f2bf(v.z); p[3] = f2bf(v.w);
  }
  __syncthreads();

  layer_pass<DIN, S0, false>(bufA, w0, biasP, bufB, nullptr, 0, quad, colQ, lane);
  __syncthreads();
  layer_pass<HID, SR, false>(bufB, wr, biasP + 768, bufA, nullptr, 0, quad, colQ, lane);
  __syncthreads();
  layer_pass<HID, SR, false>(bufA, wr + (size_t)NT * KSR * FRAG, biasP + 1536, bufB,
                             nullptr, 0, quad, colQ, lane);
  __syncthreads();
  layer_pass<HID, SR, true>(bufB, wr + (size_t)2 * NT * KSR * FRAG, biasP + 2304, nullptr,
                            out, tokenBase, quad, colQ, lane);
}

// ---------------------------------------------------------------------------
extern "C" void kernel_launch(void* const* d_in, const int* in_sizes, int n_in,
                              void* d_out, int out_size, void* d_ws, size_t ws_size,
                              hipStream_t stream) {
  const float* x   = (const float*)d_in[0];   // [32,4096,512]
  const float* W0  = (const float*)d_in[1];   // [1024,512]
  const float* Wr  = (const float*)d_in[2];   // [3,1024,256]
  const float* bih = (const float*)d_in[3];   // [4,1024]
  const float* bhh = (const float*)d_in[4];   // [4,1024]

  unsigned short* wsW0 = (unsigned short*)d_ws;                    // 48*16*512 bf16
  unsigned short* wsWr = wsW0 + (size_t)NT * KS0 * FRAG;           // 3*48*8*512 bf16
  float* biasP = (float*)(wsWr + (size_t)3 * NT * KSR * FRAG);     // 4*768 fp32

  pack_l0  <<<(NT * KS0 * FRAG) / 256, 256, 0, stream>>>(W0, wsW0);
  pack_rest<<<(3 * NT * KSR * FRAG) / 256, 256, 0, stream>>>(Wr, wsWr);
  pack_bias<<<12, 256, 0, stream>>>(bih, bhh, biasP);

  const int tokens = 32 * 4096;
  const size_t shmem = (size_t)(TOK_PER_BLK * S0 + TOK_PER_BLK * SR) * sizeof(unsigned short);
  lstm_fused<<<tokens / TOK_PER_BLK, 256, shmem, stream>>>(x, wsW0, wsWr, biasP, (float*)d_out);
}
